// myloss_62113817035047
// MI455X (gfx1250) — compile-verified, matrix operand order checked by reading, not run on hardware
//
#include <hip/hip_runtime.h>

// ---------------------------------------------------------------------------
// Types for CDNA5 WMMA
// ---------------------------------------------------------------------------
typedef float    v8f   __attribute__((ext_vector_type(8)));
typedef float    v4f   __attribute__((ext_vector_type(4)));
typedef __bf16   v16bf __attribute__((ext_vector_type(16)));
typedef __bf16   v8bf  __attribute__((ext_vector_type(8)));
typedef unsigned int v4u __attribute__((ext_vector_type(4)));

#define B_N 8192
#define D_K 256
#define DELTA_ 0.2f

// ---------------------------------------------------------------------------
// Kernel 1: row-normalize x,y -> bf16, and diag[i] = <xn_bf16[i], yn_bf16[i]>
// One wave (32 lanes) per row, 8 elements per lane.
// ---------------------------------------------------------------------------
__global__ __launch_bounds__(256) void norm_kernel(
    const float* __restrict__ x, const float* __restrict__ y,
    __bf16* __restrict__ xb, __bf16* __restrict__ yb,
    float* __restrict__ diag)
{
    const int lane = threadIdx.x & 31;
    const int row  = blockIdx.x * 8 + (threadIdx.x >> 5);

    const float* xr = x + (size_t)row * D_K + lane * 8;
    const float* yr = y + (size_t)row * D_K + lane * 8;

    v4f x0 = *(const v4f*)xr;       v4f x1 = *(const v4f*)(xr + 4);
    v4f y0 = *(const v4f*)yr;       v4f y1 = *(const v4f*)(yr + 4);

    float xf[8], yf[8];
#pragma unroll
    for (int i = 0; i < 4; ++i) { xf[i] = x0[i]; xf[i + 4] = x1[i];
                                  yf[i] = y0[i]; yf[i + 4] = y1[i]; }

    float sx = 0.f, sy = 0.f;
#pragma unroll
    for (int i = 0; i < 8; ++i) { sx = fmaf(xf[i], xf[i], sx);
                                  sy = fmaf(yf[i], yf[i], sy); }
#pragma unroll
    for (int off = 16; off > 0; off >>= 1) {
        sx += __shfl_xor(sx, off);
        sy += __shfl_xor(sy, off);
    }
    const float rix = rsqrtf(sx), riy = rsqrtf(sy);

    v8bf xo, yo;
    float dot = 0.f;
#pragma unroll
    for (int i = 0; i < 8; ++i) {
        __bf16 xv = (__bf16)(xf[i] * rix);
        __bf16 yv = (__bf16)(yf[i] * riy);
        xo[i] = xv; yo[i] = yv;
        dot = fmaf((float)xv, (float)yv, dot);
    }
    *(v8bf*)(xb + (size_t)row * D_K + lane * 8) = xo;
    *(v8bf*)(yb + (size_t)row * D_K + lane * 8) = yo;

#pragma unroll
    for (int off = 16; off > 0; off >>= 1) dot += __shfl_xor(dot, off);
    if (lane == 0) diag[row] = dot;
}

// ---------------------------------------------------------------------------
// Fragment load: 16-bit A/B fragment for v_wmma_*_16x16x32.
// Lane l<16 holds row(M)=l: K = k0+0..7 (VGPR0-3) and k0+16..23 (VGPR4-7);
// lane l+16 holds K = k0+8..15 and k0+24..31. Caller pre-adds lhalf*8.
// Both operands of the NT-GEMM (xn @ ynT) load identically row-major.
// ---------------------------------------------------------------------------
__device__ inline v16bf ldfrag(const __bf16* p) {
    union { v16bf v; struct { v4u lo; v4u hi; } u; } c;
    c.u.lo = *(const v4u*)p;            // 8 bf16 = 16B
    c.u.hi = *(const v4u*)(p + 16);     // next K-group of 8 bf16
    return c.v;
}

// ---------------------------------------------------------------------------
// Kernel 2: fused sim-tile GEMM + hinge loss + rank counting.
// Grid (64,64), block 256 = 8 waves; block tile 128x128; wave tile 64x32
// = 4(M) x 2(N) accumulators of v_wmma_f32_16x16x32_bf16 over K=256.
// ---------------------------------------------------------------------------
__global__ __launch_bounds__(256) void sim_kernel(
    const __bf16* __restrict__ xb, const __bf16* __restrict__ yb,
    const float* __restrict__ diag,
    int* __restrict__ rankG, float* __restrict__ lossG)
{
    __shared__ float diagM[128];
    __shared__ float diagN[128];
    __shared__ int   rankSh[128];
    __shared__ float lossSh;

    const int tid = threadIdx.x;
    const int bm = blockIdx.y * 128;
    const int bn = blockIdx.x * 128;

    if (tid < 128) { diagM[tid] = diag[bm + tid]; rankSh[tid] = 0; }
    else           { diagN[tid - 128] = diag[bn + tid - 128]; }
    if (tid == 0) lossSh = 0.f;
    __syncthreads();

    const int wave  = tid >> 5;
    const int lane  = tid & 31;
    const int lrow  = lane & 15;
    const int lhalf = lane >> 4;
    const int waveM = (wave >> 2) * 64;   // 0 or 64
    const int waveN = (wave & 3) * 32;    // 0,32,64,96

    v8f acc[4][2];
#pragma unroll
    for (int mi = 0; mi < 4; ++mi)
#pragma unroll
        for (int ni = 0; ni < 2; ++ni) acc[mi][ni] = (v8f)0.f;

    const __bf16* aBase = xb + (size_t)(bm + waveM + lrow) * D_K + lhalf * 8;
    const __bf16* bBase = yb + (size_t)(bn + waveN + lrow) * D_K + lhalf * 8;

    // K loop: EXEC all-ones, no divergence (WMMA requirement).
#pragma unroll 2
    for (int k0 = 0; k0 < D_K; k0 += 32) {
        v16bf a[4], b[2];
#pragma unroll
        for (int mi = 0; mi < 4; ++mi)
            a[mi] = ldfrag(aBase + (size_t)mi * 16 * D_K + k0);
#pragma unroll
        for (int ni = 0; ni < 2; ++ni)
            b[ni] = ldfrag(bBase + (size_t)ni * 16 * D_K + k0);
#pragma unroll
        for (int mi = 0; mi < 4; ++mi)
#pragma unroll
            for (int ni = 0; ni < 2; ++ni)
                acc[mi][ni] = __builtin_amdgcn_wmma_f32_16x16x32_bf16(
                    false, a[mi], false, b[ni], (short)0, acc[mi][ni],
                    false, false);
    }

    // Epilogue: C/D layout: lane<16 -> (M=v, N=lane); lane>=16 -> (M=8+v, N=lane-16)
    float lossLocal = 0.f;
#pragma unroll
    for (int mi = 0; mi < 4; ++mi) {
        const int rowB = waveM + mi * 16;
#pragma unroll
        for (int ni = 0; ni < 2; ++ni) {
            const int colL = waveN + ni * 16 + lrow;
            const float dj = diagN[colL];
            const int gj = bn + colL;
#pragma unroll
            for (int v = 0; v < 8; ++v) {
                const int rowL = rowB + lhalf * 8 + v;
                const float s  = acc[mi][ni][v];
                const float di = diagM[rowL];
                const bool nd  = (bm + rowL) != gj;   // off-diagonal
                const float t  = fmaxf(s - di + DELTA_, 0.f)
                               + fmaxf(s - dj + DELTA_, 0.f);
                lossLocal += nd ? t : 0.f;
                // rank count: low 16 lanes -> row rowB+v, high -> rowB+8+v
                unsigned msk = (unsigned)__ballot(nd && (s > di));
                if (lane == 0) {
                    int c0 = __popc(msk & 0xffffu);
                    int c1 = __popc(msk >> 16);
                    if (c0) atomicAdd(&rankSh[rowB + v], c0);
                    if (c1) atomicAdd(&rankSh[rowB + 8 + v], c1);
                }
            }
        }
    }

    atomicAdd(&lossSh, lossLocal);
    __syncthreads();

    if (tid < 128) {
        int c = rankSh[tid];
        if (c) atomicAdd(&rankG[bm + tid], c);
    }
    if (tid == 0) atomicAdd(lossG, lossSh);
}

// ---------------------------------------------------------------------------
// Kernel 3: histogram the diagonal ranks -> loss / recall / ndcg (7 floats)
// ---------------------------------------------------------------------------
__global__ void finalize_kernel(const int* __restrict__ rank,
                                const float* __restrict__ lossAcc,
                                float* __restrict__ out)
{
    __shared__ int hist[10];
    const int tid = threadIdx.x;
    if (tid < 10) hist[tid] = 0;
    __syncthreads();
    for (int i = tid; i < B_N; i += 256) {
        int r = rank[i];
        if (r < 10) atomicAdd(&hist[r], 1);
    }
    __syncthreads();
    if (tid == 0) {
        out[0] = lossAcc[0] / (float)B_N;
        int cum = 0; float nd = 0.f;
        for (int r = 0; r < 10; ++r) {
            float wr = 0.69314718055994531f / logf(2.0f + (float)r);
            cum += hist[r];
            nd  += (float)hist[r] * wr;
            if (r == 0) { out[1] = (float)cum; out[4] = nd; }
            if (r == 4) { out[2] = (float)cum; out[5] = nd; }
            if (r == 9) { out[3] = (float)cum; out[6] = nd; }
        }
    }
}

// ---------------------------------------------------------------------------
// Launch
// ---------------------------------------------------------------------------
extern "C" void kernel_launch(void* const* d_in, const int* in_sizes, int n_in,
                              void* d_out, int out_size, void* d_ws, size_t ws_size,
                              hipStream_t stream)
{
    (void)in_sizes; (void)n_in; (void)out_size; (void)ws_size;
    const float* x = (const float*)d_in[0];
    const float* y = (const float*)d_in[1];
    float* out = (float*)d_out;

    char* ws = (char*)d_ws;
    const size_t XB_BYTES = (size_t)B_N * D_K * 2;      // 4 MiB each
    __bf16* xb   = (__bf16*)(ws);
    __bf16* yb   = (__bf16*)(ws + XB_BYTES);
    float*  diag = (float*) (ws + 2 * XB_BYTES);
    int*    rank = (int*)   (ws + 2 * XB_BYTES + (size_t)B_N * 4);
    float*  lossG= (float*) (ws + 2 * XB_BYTES + (size_t)B_N * 8);

    // zero rank[] and loss accumulator (contiguous) each call
    hipMemsetAsync(rank, 0, (size_t)B_N * 4 + 64, stream);

    norm_kernel<<<B_N / 8, 256, 0, stream>>>(x, y, xb, yb, diag);
    sim_kernel<<<dim3(B_N / 128, B_N / 128), 256, 0, stream>>>(xb, yb, diag, rank, lossG);
    finalize_kernel<<<1, 256, 0, stream>>>(rank, lossG, out);
}
